// ChessStructureMP_76373108457571
// MI455X (gfx1250) — compile-verified, matrix operand order checked by reading, not run on hardware
//
#include <hip/hip_runtime.h>

// ---------------------------------------------------------------------------
// Types for CDNA5 WMMA (wave32).
// ---------------------------------------------------------------------------
typedef __attribute__((ext_vector_type(16))) __bf16 v16bf;
typedef __attribute__((ext_vector_type(8)))  __bf16 v8bf;
typedef __attribute__((ext_vector_type(8)))  float  v8f;

#define B_N   2048
#define S_N   64
#define D_N   512
#define H_N   8
#define HD_N  64
#define MROWS (B_N * S_N)           // 131072
#define NQKV  (3 * D_N)             // 1536
#define SCALE 0.125f                // 1/sqrt(64)

// Load one 16x32 bf16 WMMA fragment from a row-major (K-contiguous) buffer.
// Per ISA 7.12.2: lane L holds row (L&15); K chunks at 8*(L>>4) and +16.
// Caller passes p already offset to [row*stride + 8*(lane>>4)].
__device__ __forceinline__ v16bf frag16(const __bf16* p) {
    union { v16bf v; v8bf h[2]; } u;
    u.h[0] = *(const v8bf*)(p);
    u.h[1] = *(const v8bf*)(p + 16);
    return u.v;
}

__device__ __forceinline__ v8f wmma_bf16(v16bf a, v16bf b, v8f c) {
    return __builtin_amdgcn_wmma_f32_16x16x32_bf16(
        false, a, false, b, (short)0, c, false, false);
}

// ---------------------------------------------------------------------------
// Kernel 0: pack Wq/Wk/Wv -> stacked bf16 (1536x512), Wo -> bf16, biases -> f32
// ---------------------------------------------------------------------------
__global__ __launch_bounds__(256) void prep_weights(
    const float* __restrict__ Wq, const float* __restrict__ Wk,
    const float* __restrict__ Wv, const float* __restrict__ Wo,
    const float* __restrict__ bq, const float* __restrict__ bk,
    const float* __restrict__ bv,
    __bf16* __restrict__ Wqkv, __bf16* __restrict__ Wob,
    float* __restrict__ bqkv) {
    int tid = blockIdx.x * 256 + threadIdx.x;      // 4096 blocks -> 1,048,576
    const int WSZ = D_N * D_N;                      // 262144
    if (tid < 3 * WSZ) {
        int sel = tid / WSZ;
        int off = tid - sel * WSZ;
        const float* W = (sel == 0) ? Wq : (sel == 1 ? Wk : Wv);
        Wqkv[tid] = (__bf16)W[off];
    } else {
        int off = tid - 3 * WSZ;
        Wob[off] = (__bf16)Wo[off];
    }
    if (tid < NQKV) {
        float v;
        if (tid < D_N)            v = bq[tid];
        else if (tid < 2 * D_N)   v = bk[tid - D_N];
        else                      v = bv[tid - 2 * D_N];
        bqkv[tid] = v;
    }
}

// ---------------------------------------------------------------------------
// Kernel 1/3: Y = A * W^T + bias.  A: MxK (f32 or bf16), W: NxK (bf16, row
// major, K contiguous -> natural WMMA B operand).  128x128 WG tile, BK=32,
// 8 waves, each wave owns a 64x32 sub-tile (4x2 accumulators).
// Software pipeline: prefetch next K-tile into VGPRs, double-buffered LDS,
// one barrier per K-step; WMMAs overlap the in-flight global loads.
// ---------------------------------------------------------------------------
template<int A_IS_BF16, int OUT_BF16>
__global__ __launch_bounds__(256) void gemm_bias(
    const void* __restrict__ Aptr, const __bf16* __restrict__ Bw,
    const float* __restrict__ bias, void* __restrict__ Out,
    int M, int N, int K) {
    // padded stride 40 bf16 (=80B, multiple of 16B) keeps b128 frag loads aligned
    __shared__ __bf16 sA[2][128 * 40];
    __shared__ __bf16 sB[2][128 * 40];

    const int tid    = threadIdx.x;
    const int lane   = tid & 31;
    const int wave   = tid >> 5;
    const int lane15 = lane & 15;
    const int kb     = (lane >> 4) << 3;   // 0 or 8
    const int mw     = (wave >> 2) * 64;   // wave M offset in tile
    const int nw     = (wave & 3) * 32;    // wave N offset in tile
    const int n0     = blockIdx.x * 128;   // N fast -> A slab shared via L2
    const int m0     = blockIdx.y * 128;

    // per-thread staging coordinates
    // f32 A path: 4 x float4  (row = idx>>3, kc = (idx&7)*4)
    // bf16 path : 2 x v8bf    (row = idx>>2, kc = (idx&3)*8)
    int rowF[4], kcF[4], rowH[2], kcH[2];
    #pragma unroll
    for (int i = 0; i < 4; ++i) {
        int idx = i * 256 + tid;
        rowF[i] = idx >> 3; kcF[i] = (idx & 7) << 2;
    }
    #pragma unroll
    for (int i = 0; i < 2; ++i) {
        int idx = i * 256 + tid;
        rowH[i] = idx >> 2; kcH[i] = (idx & 3) << 3;
    }

    float4 ra[4];   // A prefetch (f32 path)
    v8bf   rah[2];  // A prefetch (bf16 path)
    v8bf   rb[2];   // B prefetch

    auto load_tiles = [&](int k0) {
        if (A_IS_BF16) {
            const __bf16* A = (const __bf16*)Aptr;
            #pragma unroll
            for (int i = 0; i < 2; ++i)
                rah[i] = *(const v8bf*)(A + (size_t)(m0 + rowH[i]) * K + k0 + kcH[i]);
        } else {
            const float* A = (const float*)Aptr;
            #pragma unroll
            for (int i = 0; i < 4; ++i)
                ra[i] = *(const float4*)(A + (size_t)(m0 + rowF[i]) * K + k0 + kcF[i]);
        }
        #pragma unroll
        for (int i = 0; i < 2; ++i)
            rb[i] = *(const v8bf*)(Bw + (size_t)(n0 + rowH[i]) * K + k0 + kcH[i]);
    };
    auto store_tiles = [&](int buf) {
        if (A_IS_BF16) {
            #pragma unroll
            for (int i = 0; i < 2; ++i)
                *(v8bf*)&sA[buf][rowH[i] * 40 + kcH[i]] = rah[i];
        } else {
            #pragma unroll
            for (int i = 0; i < 4; ++i) {
                union { __bf16 h[4]; uint2 u; } pk;
                pk.h[0] = (__bf16)ra[i].x; pk.h[1] = (__bf16)ra[i].y;
                pk.h[2] = (__bf16)ra[i].z; pk.h[3] = (__bf16)ra[i].w;
                *(uint2*)&sA[buf][rowF[i] * 40 + kcF[i]] = pk.u;   // 8B aligned
            }
        }
        #pragma unroll
        for (int i = 0; i < 2; ++i)
            *(v8bf*)&sB[buf][rowH[i] * 40 + kcH[i]] = rb[i];
    };

    v8f acc[4][2] = {};

    load_tiles(0);
    int buf = 0;
    for (int k0 = 0; k0 < K; k0 += 32) {
        store_tiles(buf);
        __syncthreads();
        if (k0 + 32 < K) load_tiles(k0 + 32);   // in-flight during WMMAs

        v16bf af[4], bfr[2];
        #pragma unroll
        for (int mt = 0; mt < 4; ++mt)
            af[mt] = frag16(&sA[buf][(mw + mt * 16 + lane15) * 40 + kb]);
        #pragma unroll
        for (int nt = 0; nt < 2; ++nt)
            bfr[nt] = frag16(&sB[buf][(nw + nt * 16 + lane15) * 40 + kb]);
        #pragma unroll
        for (int mt = 0; mt < 4; ++mt)
            #pragma unroll
            for (int nt = 0; nt < 2; ++nt)
                acc[mt][nt] = wmma_bf16(af[mt], bfr[nt], acc[mt][nt]);
        buf ^= 1;
    }

    // ---- epilogue: bias + store.  C/D layout: m = r + 8*(lane>>4), n = lane&15
    float bval[2]; int cols[2];
    #pragma unroll
    for (int nt = 0; nt < 2; ++nt) {
        cols[nt] = n0 + nw + nt * 16 + lane15;
        bval[nt] = bias[cols[nt]];
    }
    const int mtop = m0 + mw + ((lane >> 4) << 3);
    #pragma unroll
    for (int mt = 0; mt < 4; ++mt)
        #pragma unroll
        for (int nt = 0; nt < 2; ++nt)
            #pragma unroll
            for (int r = 0; r < 8; ++r) {
                int rowg = mtop + mt * 16 + r;
                float v = acc[mt][nt][r] + bval[nt];
                if (OUT_BF16)
                    ((__bf16*)Out)[(size_t)rowg * N + cols[nt]] = (__bf16)v;
                else
                    ((float*)Out)[(size_t)rowg * N + cols[nt]] = v;
            }
}

// ---------------------------------------------------------------------------
// Kernel 2: masked attention, one workgroup per (batch, head).
// qkv layout: [(b*64+s)*1536 + {0|512|1024} + h*64 + hd]  (bf16)
// ---------------------------------------------------------------------------
__global__ __launch_bounds__(128) void attn_kernel(
    const __bf16* __restrict__ qkv,
    const unsigned char* __restrict__ ray,
    const unsigned char* __restrict__ atk,
    __bf16* __restrict__ aout) {
    __shared__ __bf16 sVT[64 * 80];   // v transposed: [hd][t], stride 80 (160B)
    __shared__ float  sS [64 * 64];   // scores / exp scratch
    __shared__ __bf16 sP [64 * 80];   // probabilities, stride 80

    const int bh   = blockIdx.x;
    const int b    = bh >> 3;
    const int h    = bh & 7;
    const int tid  = threadIdx.x;
    const int lane = tid & 31;
    const int wave = tid >> 5;
    const int lane15 = lane & 15;
    const int kb     = (lane >> 4) << 3;
    const size_t qbase = (size_t)b * S_N * NQKV + h * HD_N;

    // ---- stage v transposed into LDS: b128 global loads, scatter on LDS side
    #pragma unroll
    for (int i = 0; i < 4; ++i) {
        int c   = i * 128 + tid;          // 512 chunks of 8 bf16
        int t   = c >> 3;
        int hd0 = (c & 7) << 3;
        v8bf v = *(const v8bf*)(qkv + qbase + 2 * D_N + (size_t)t * NQKV + hd0);
        #pragma unroll
        for (int j = 0; j < 8; ++j)
            sVT[(hd0 + j) * 80 + t] = v[j];
    }

    // ---- scores = q @ k^T : wave w owns query rows [16w, 16w+16) ----
    const int mrow = wave * 16 + lane15;
    v8f acc[4] = {};
    #pragma unroll
    for (int kc = 0; kc < HD_N; kc += 32) {
        v16bf af = frag16(qkv + qbase + (size_t)mrow * NQKV + kc + kb);
        #pragma unroll
        for (int nt = 0; nt < 4; ++nt) {
            int trow = nt * 16 + lane15;
            v16bf bfr = frag16(qkv + qbase + D_N + (size_t)trow * NQKV + kc + kb);
            acc[nt] = wmma_bf16(af, bfr, acc[nt]);
        }
    }
    const int mloc = wave * 16 + ((lane >> 4) << 3);
    #pragma unroll
    for (int nt = 0; nt < 4; ++nt)
        #pragma unroll
        for (int r = 0; r < 8; ++r)
            sS[(mloc + r) * 64 + nt * 16 + lane15] = acc[nt][r] * SCALE;
    __syncthreads();

    // ---- masked softmax: a lane PAIR per query row (wave32 shuffles) ----
    {
        const int m    = tid >> 1;            // row
        const int c0   = (tid & 1) << 5;      // this lane's 32-col half
        const int ri = m >> 3, fi = m & 7;
        float mx = -3.0e38f;
        #pragma unroll 4
        for (int j = 0; j < 32; ++j) {
            int t = c0 + j;
            int rj = t >> 3, fj = t & 7;
            int dr = (ri > rj) ? ri - rj : rj - ri;
            int df = (fi > fj) ? fi - fj : fj - fi;
            bool keep;
            if      (h == 0) keep = (fi == fj);
            else if (h == 1) keep = (ri == rj);
            else if (h == 2) keep = (ri - fi) == (rj - fj);
            else if (h == 3) keep = (ri + fi) == (rj + fj);
            else if (h == 4) keep = ((dr == 2 && df == 1) || (dr == 1 && df == 2) || (m == t));
            else if (h == 5) keep = (dr <= 1 && df <= 1);
            else if (h == 6) keep = (ray[((size_t)b << 12) + (m << 6) + t] != 0) || (m == t);
            else             keep = (atk[((size_t)b << 12) + (m << 6) + t] != 0) || (m == t);
            float v = keep ? sS[m * 64 + t] : -1.0e9f;
            sS[m * 64 + t] = v;
            mx = fmaxf(mx, v);
        }
        mx = fmaxf(mx, __shfl_xor(mx, 1));    // combine lane pair
        float sum = 0.0f;
        #pragma unroll 4
        for (int j = 0; j < 32; ++j) {
            float e = __expf(sS[m * 64 + c0 + j] - mx);
            sS[m * 64 + c0 + j] = e;
            sum += e;
        }
        sum += __shfl_xor(sum, 1);
        float inv = 1.0f / sum;
        #pragma unroll 4
        for (int j = 0; j < 32; ++j)
            sP[m * 80 + c0 + j] = (__bf16)(sS[m * 64 + c0 + j] * inv);
    }
    __syncthreads();

    // ---- out = attn @ v ----
    v8f oc[4] = {};
    #pragma unroll
    for (int kc = 0; kc < S_N; kc += 32) {
        v16bf af = frag16(&sP[mrow * 80 + kc + kb]);
        #pragma unroll
        for (int nt = 0; nt < 4; ++nt) {
            v16bf bfr = frag16(&sVT[(nt * 16 + lane15) * 80 + kc + kb]);
            oc[nt] = wmma_bf16(af, bfr, oc[nt]);
        }
    }
    #pragma unroll
    for (int nt = 0; nt < 4; ++nt)
        #pragma unroll
        for (int r = 0; r < 8; ++r) {
            int m  = mloc + r;
            int hd = nt * 16 + lane15;
            aout[((size_t)b * S_N + m) * D_N + h * HD_N + hd] = (__bf16)oc[nt][r];
        }
}

// ---------------------------------------------------------------------------
// Host launcher
// ---------------------------------------------------------------------------
extern "C" void kernel_launch(void* const* d_in, const int* in_sizes, int n_in,
                              void* d_out, int out_size, void* d_ws, size_t ws_size,
                              hipStream_t stream) {
    const float* x  = (const float*)d_in[0];
    const unsigned char* ray = (const unsigned char*)d_in[1];
    const unsigned char* atk = (const unsigned char*)d_in[2];
    const float* Wq = (const float*)d_in[3];
    const float* bq = (const float*)d_in[4];
    const float* Wk = (const float*)d_in[5];
    const float* bk = (const float*)d_in[6];
    const float* Wv = (const float*)d_in[7];
    const float* bv = (const float*)d_in[8];
    const float* Wo = (const float*)d_in[9];
    const float* bo = (const float*)d_in[10];

    char* ws = (char*)d_ws;
    size_t off = 0;
    __bf16* Wqkv = (__bf16*)(ws + off); off += (size_t)NQKV * D_N * 2;      // 1.5 MB
    float*  bqkv = (float*) (ws + off); off += 8192;                        // biases (padded)
    __bf16* Wob  = (__bf16*)(ws + off); off += (size_t)D_N * D_N * 2;       // 0.5 MB
    __bf16* qkv  = (__bf16*)(ws + off); off += (size_t)MROWS * NQKV * 2;    // 384 MB
    __bf16* aattn= (__bf16*)(ws + off); off += (size_t)MROWS * D_N * 2;     // 128 MB

    prep_weights<<<4096, 256, 0, stream>>>(Wq, Wk, Wv, Wo, bq, bk, bv,
                                           Wqkv, Wob, bqkv);

    // QKV: M=131072, N=1536, K=512 ; grid.x = N tiles (fast) so the 12 WGs
    // of one M-slab run together and share the x slab through L2.
    gemm_bias<0, 1><<<dim3(NQKV / 128, MROWS / 128), 256, 0, stream>>>(
        (const void*)x, Wqkv, bqkv, (void*)qkv, MROWS, NQKV, D_N);

    attn_kernel<<<B_N * H_N, 128, 0, stream>>>(qkv, ray, atk, aattn);

    gemm_bias<1, 0><<<dim3(D_N / 128, MROWS / 128), 256, 0, stream>>>(
        (const void*)aattn, Wob, bo, d_out, MROWS, D_N, D_N);
}
